// CausalSelfAttention_11751030522050
// MI455X (gfx1250) — compile-verified, matrix operand order checked by reading, not run on hardware
//
#include <hip/hip_runtime.h>
#include <type_traits>

// ---------------------------------------------------------------------------
// CDNA5 (gfx1250, wave32) causal self-attention:
//   qkv = x @ W_attn  -> RoPE(q,k) -> flash attention -> y @ W_proj
// Matmuls: v_wmma_f32_16x16x32_bf16 (fp32 accum). A-tile staging in the
// bf16-A GEMM uses the Tensor Data Mover (tensor_load_to_lds + TENSORcnt).
// ---------------------------------------------------------------------------

typedef __attribute__((ext_vector_type(8)))  __bf16 v8bf;
typedef __attribute__((ext_vector_type(16))) __bf16 v16bf;
typedef __attribute__((ext_vector_type(8)))  float  v8f;
typedef __attribute__((ext_vector_type(4)))  unsigned int u32x4;
typedef __attribute__((ext_vector_type(8)))  int i32x8;
typedef __attribute__((ext_vector_type(4)))  int i32x4;

#define WMMA_BF16(A, B, Cc) \
  __builtin_amdgcn_wmma_f32_16x16x32_bf16(false, (A), false, (B), (short)0, (Cc), false, false)

#if __has_builtin(__builtin_amdgcn_tensor_load_to_lds) && \
    __has_builtin(__builtin_amdgcn_s_wait_tensorcnt)
#define USE_TDM 1
#else
#define USE_TDM 0
#endif

__device__ __forceinline__ v16bf make_frag(const __bf16* p0, const __bf16* p1) {
  v8bf lo = *(const v8bf*)p0;
  v8bf hi = *(const v8bf*)p1;
  return __builtin_shufflevector(lo, hi, 0, 1, 2, 3, 4, 5, 6, 7,
                                 8, 9, 10, 11, 12, 13, 14, 15);
}

__device__ __forceinline__ v8f v8f_zero() {
  v8f z;
#pragma unroll
  for (int i = 0; i < 8; ++i) z[i] = 0.0f;
  return z;
}

__device__ __forceinline__ float redmax16(float v) {
#pragma unroll
  for (int m = 1; m < 16; m <<= 1) v = fmaxf(v, __shfl_xor(v, m, 16));
  return v;
}
__device__ __forceinline__ float redsum16(float v) {
#pragma unroll
  for (int m = 1; m < 16; m <<= 1) v += __shfl_xor(v, m, 16);
  return v;
}

// ---------------------------------------------------------------------------
// Generic WMMA GEMM: C[M,N] = A[M,K] * Bw[K,N]  (Bw fp32 -> bf16 in LDS)
// Block tile 128x128, BK=32, 256 threads = 8 waves (2x4), wave tile 64x32.
// LDS stride 40 bf16 (80 B): bank-conflict-friendly, 16B aligned for b128.
// bf16-A path: A tile staged by the Tensor Data Mover (2D tile descriptor,
// pad_interval=16 DW / pad_amount=4 DW reproduces the 40-element stride).
// ---------------------------------------------------------------------------
template <typename TA, typename TC>
__global__ __launch_bounds__(256) void gemm_wmma_kernel(
    const TA* __restrict__ A, const float* __restrict__ Bw, TC* __restrict__ Cc,
    int M, int N, int K) {
  __shared__ __bf16 As[128 * 40];
  __shared__ __bf16 Bs[128 * 40];

  const int tid  = threadIdx.x;
  const int lane = tid & 31;
  const int wid  = tid >> 5;
  const int half = lane >> 4;   // which 16-lane half of the wave32
  const int r    = lane & 15;
  const int wm   = wid >> 2;    // 0..1
  const int wn   = wid & 3;     // 0..3
  const size_t m0 = (size_t)blockIdx.y * 128;
  const size_t n0 = (size_t)blockIdx.x * 128;

  v8f acc[4][2];
#pragma unroll
  for (int mf = 0; mf < 4; ++mf)
#pragma unroll
    for (int nf = 0; nf < 2; ++nf) acc[mf][nf] = v8f_zero();

  for (int k0 = 0; k0 < K; k0 += 32) {
    __syncthreads();

    // prefetch next K-tile while this one is staged/computed
    if (k0 + 32 < K) {
      if (tid < 128) __builtin_prefetch(A + (m0 + tid) * K + k0 + 32, 0, 1);
      __builtin_prefetch(Bw + (size_t)(k0 + 32 + (tid >> 3)) * N + n0 + (tid & 7) * 16, 0, 1);
    }

    // ---- stage A tile (128 x 32) into LDS as bf16 ----
    bool tdm_issued = false;
    if constexpr (std::is_same_v<TA, float>) {
#pragma unroll
      for (int it = 0; it < 4; ++it) {
        const int lin = tid + it * 256;           // 0..1023 float4 chunks
        const int row = lin >> 3;
        const int kc  = (lin & 7) * 4;
        const float4 f = *(const float4*)(A + (m0 + row) * K + k0 + kc);
        __bf16* d = As + row * 40 + kc;
        d[0] = (__bf16)f.x; d[1] = (__bf16)f.y;
        d[2] = (__bf16)f.z; d[3] = (__bf16)f.w;
      }
    } else {
#if USE_TDM
      // TDM: 128x32 bf16 tile, row length 64B (16 DW), then 16B (4 DW) pad.
      if (wid == 0) {
        const unsigned long long ga =
            (unsigned long long)(uintptr_t)(A + m0 * K + k0);
        const unsigned lds_off = (unsigned)(uintptr_t)(void*)As;
        u32x4 g0;
        g0[0] = 1u;                                    // count=1 (valid user D#)
        g0[1] = lds_off;                               // lds_addr
        g0[2] = (unsigned)(ga & 0xffffffffu);          // global_addr[31:0]
        g0[3] = (unsigned)((ga >> 32) & 0x01ffffffu)   // global_addr[56:32]
              | (2u << 30);                            // type = 2 ("image")
        i32x8 g1;
        g1[0] = (1 << 16)      // data_size = 2 bytes
              | (1 << 20)      // pad_enable
              | (3 << 22)      // pad_interval: 16 DWORDs (= 32 bf16 row)
              | (3 << 25);     // pad_amount: 4 DWORDs (= 8 bf16)
        g1[1] = (int)(((unsigned)K & 0xffffu) << 16);                 // dim0 lo
        g1[2] = (int)((((unsigned)K >> 16) & 0xffffu)
              | (((unsigned)M & 0xffffu) << 16));                     // dim0 hi | dim1 lo
        g1[3] = (int)((((unsigned)M >> 16) & 0xffffu) | (32u << 16)); // dim1 hi | tile_dim0=32
        g1[4] = (int)128u;                                            // tile_dim1=128
        g1[5] = (int)(unsigned)K;                                     // dim0_stride lo
        g1[6] = 0;
        g1[7] = 0;
        i32x4 z4;
        z4[0] = z4[1] = z4[2] = z4[3] = 0;
        i32x8 z8;
#pragma unroll
        for (int zi = 0; zi < 8; ++zi) z8[zi] = 0;
        // 6-arg form (clang-23 / therock-10.0 headers): groups 0..3 + spare + cpol
        __builtin_amdgcn_tensor_load_to_lds(g0, g1, z4, z4, z8, 0);
        tdm_issued = true;
      }
#else
#pragma unroll
      for (int it = 0; it < 2; ++it) {
        const int lin = tid + it * 256;           // 0..511 v8bf chunks
        const int row = lin >> 2;
        const int kc  = (lin & 3) * 8;
        *(v8bf*)(As + row * 40 + kc) =
            *(const v8bf*)(A + (m0 + row) * K + k0 + kc);
      }
#endif
    }

    // ---- stage B tile (32 x 128) transposed -> Bs[n][k], bf16 ----
#pragma unroll
    for (int it = 0; it < 4; ++it) {
      const int lin = tid + it * 256;             // 0..1023 float4 chunks
      const int kr  = lin >> 5;                   // 0..31
      const int nc  = (lin & 31) * 4;
      const float4 f = *(const float4*)(Bw + (size_t)(k0 + kr) * N + n0 + nc);
      Bs[(nc + 0) * 40 + kr] = (__bf16)f.x;
      Bs[(nc + 1) * 40 + kr] = (__bf16)f.y;
      Bs[(nc + 2) * 40 + kr] = (__bf16)f.z;
      Bs[(nc + 3) * 40 + kr] = (__bf16)f.w;
    }
#if USE_TDM
    if (tdm_issued) __builtin_amdgcn_s_wait_tensorcnt(0);
#else
    (void)tdm_issued;
#endif
    __syncthreads();

    // ---- fragments + 8 WMMAs ----
    v16bf af[4], bfr[2];
#pragma unroll
    for (int mf = 0; mf < 4; ++mf) {
      const __bf16* p = As + (wm * 64 + mf * 16 + r) * 40 + half * 8;
      af[mf] = make_frag(p, p + 16);
    }
#pragma unroll
    for (int nf = 0; nf < 2; ++nf) {
      const __bf16* p = Bs + (wn * 32 + nf * 16 + r) * 40 + half * 8;
      bfr[nf] = make_frag(p, p + 16);
    }
#pragma unroll
    for (int mf = 0; mf < 4; ++mf)
#pragma unroll
      for (int nf = 0; nf < 2; ++nf)
        acc[mf][nf] = WMMA_BF16(af[mf], bfr[nf], acc[mf][nf]);
  }

  // ---- epilogue: C-frag row = i + half*8, col = lane%16 ----
#pragma unroll
  for (int mf = 0; mf < 4; ++mf) {
    const size_t row0 = m0 + wm * 64 + mf * 16 + half * 8;
#pragma unroll
    for (int nf = 0; nf < 2; ++nf) {
      const size_t col = n0 + wn * 32 + nf * 16 + r;
#pragma unroll
      for (int i = 0; i < 8; ++i)
        Cc[(row0 + i) * N + col] = (TC)acc[mf][nf][i];
    }
  }
}

// ---------------------------------------------------------------------------
// RoPE + head-major repack. One thread per (b,t,h,i), i in [0,32) = half HD.
// q gets 1/sqrt(HD)=0.125 pre-folded. v stored transposed [B,H,HD,T] so the
// P@V WMMA B-fragments (per-lane contiguous K = k rows) are vector loads.
// ---------------------------------------------------------------------------
__global__ __launch_bounds__(256) void rope_pack_kernel(
    const __bf16* __restrict__ qkv, const int* __restrict__ idx,
    __bf16* __restrict__ qb, __bf16* __restrict__ kb, __bf16* __restrict__ vt) {
  constexpr int T = 2048, H = 16, HD = 64, C3 = 3072;
  const int gid = blockIdx.x * 256 + threadIdx.x;   // 2^22 threads
  const int i = gid & 31;
  const int h = (gid >> 5) & 15;
  const int t = (gid >> 9) & (T - 1);
  const int b = gid >> 20;

  const float pos = (float)idx[b * T + t];
  const float inv = __expf(-(float)i * 0.28782313662425572f);  // ln(1e4)/32
  float sn, cs;
  __sincosf(pos * inv, &sn, &cs);

  const size_t base = (size_t)(b * T + t) * C3 + h * HD;
  const float q1 = (float)qkv[base + i],        q2 = (float)qkv[base + 32 + i];
  const float k1 = (float)qkv[base + 1024 + i], k2 = (float)qkv[base + 1056 + i];
  const float v1 = (float)qkv[base + 2048 + i], v2 = (float)qkv[base + 2080 + i];

  const size_t hb = ((size_t)(b * H + h) * T + t) * HD;
  qb[hb + i]      = (__bf16)((q1 * cs - q2 * sn) * 0.125f);
  qb[hb + 32 + i] = (__bf16)((q1 * sn + q2 * cs) * 0.125f);
  kb[hb + i]      = (__bf16)(k1 * cs - k2 * sn);
  kb[hb + 32 + i] = (__bf16)(k1 * sn + k2 * cs);
  const size_t vb = (size_t)(b * H + h) * HD;
  vt[(vb + i)      * T + t] = (__bf16)v1;
  vt[(vb + 32 + i) * T + t] = (__bf16)v2;
}

// ---------------------------------------------------------------------------
// Flash attention: 128 threads = 4 waves; each wave owns a 16-row q tile and
// streams 32-wide k tiles up to the causal limit. S via 2 chained WMMAs over
// HD=64; online softmax with 16-lane shuffles; interior tiles (k0+31 <= q0)
// take the mask-free path (wave-uniform branch). P is round-tripped through
// a private LDS region (C-layout -> A-layout, s_wait_dscnt 0 covers the RAW).
// ---------------------------------------------------------------------------
__global__ __launch_bounds__(128) void attn_kernel(
    const __bf16* __restrict__ qb, const __bf16* __restrict__ kb,
    const __bf16* __restrict__ vt, __bf16* __restrict__ yb) {
  constexpr int T = 2048, HD = 64, C = 1024;
  __shared__ __bf16 Plds[4][16 * 40];

  const int tid  = threadIdx.x;
  const int wid  = tid >> 5;
  const int lane = tid & 31;
  const int half = lane >> 4;
  const int r    = lane & 15;

  const int bh = blockIdx.y;           // b*H + h
  const int b  = bh >> 4;
  const int h  = bh & 15;
  const int q0 = blockIdx.x * 64 + wid * 16;

  const __bf16* qptr = qb + ((size_t)bh * T + q0) * HD;
  const __bf16* kptr = kb + (size_t)bh * T * HD;
  const __bf16* vptr = vt + (size_t)bh * HD * T;

  v16bf qf[2];
#pragma unroll
  for (int s = 0; s < 2; ++s) {
    const __bf16* p = qptr + (size_t)r * HD + s * 32 + half * 8;
    qf[s] = make_frag(p, p + 16);
  }

  v8f o[4];
#pragma unroll
  for (int nf = 0; nf < 4; ++nf) o[nf] = v8f_zero();
  float mrow[8], lrow[8], alpha[8];
#pragma unroll
  for (int i = 0; i < 8; ++i) { mrow[i] = -__builtin_inff(); lrow[i] = 0.0f; }

  const int kmax = q0 + 15;
  for (int k0 = 0; k0 <= kmax; k0 += 32) {
    // ---- S = q @ k^T for two 16-col sub-tiles ----
    v8f sa = v8f_zero(), sb = v8f_zero();
#pragma unroll
    for (int s = 0; s < 2; ++s) {
      const __bf16* p0 = kptr + (size_t)(k0 + r) * HD + s * 32 + half * 8;
      sa = WMMA_BF16(qf[s], make_frag(p0, p0 + 16), sa);
      const __bf16* p1 = kptr + (size_t)(k0 + 16 + r) * HD + s * 32 + half * 8;
      sb = WMMA_BF16(qf[s], make_frag(p1, p1 + 16), sb);
    }

    // ---- online softmax; only diagonal tiles pay for the causal mask ----
    __bf16* pw = &Plds[wid][0];
    auto softmax_update = [&](auto masked_t) {
      constexpr bool MASKED = decltype(masked_t)::value;
#pragma unroll
      for (int i = 0; i < 8; ++i) {
        float e0, e1;
        if constexpr (MASKED) {
          const int row = q0 + i + half * 8;
          e0 = (k0 + r      <= row) ? sa[i] : -__builtin_inff();
          e1 = (k0 + 16 + r <= row) ? sb[i] : -__builtin_inff();
        } else {
          e0 = sa[i];
          e1 = sb[i];
        }
        float mn = redmax16(fmaxf(e0, e1));
        mn = fmaxf(mrow[i], mn);
        const float al = __expf(mrow[i] - mn);
        const float p0 = __expf(e0 - mn);
        const float p1 = __expf(e1 - mn);
        lrow[i] = lrow[i] * al + redsum16(p0 + p1);
        mrow[i] = mn;
        alpha[i] = al;
        pw[(i + half * 8) * 40 + r]      = (__bf16)p0;
        pw[(i + half * 8) * 40 + r + 16] = (__bf16)p1;
      }
    };
    if (k0 + 31 > q0) softmax_update(std::true_type{});   // wave-uniform branch
    else              softmax_update(std::false_type{});

    asm volatile("s_wait_dscnt 0" ::: "memory");  // LDS RAW: P store -> A-frag load

    const __bf16* pr = pw + r * 40 + half * 8;
    const v16bf pa = make_frag(pr, pr + 16);      // P as 16x32 A-fragment

    // ---- O = diag(alpha) * O + P @ V ----
#pragma unroll
    for (int nf = 0; nf < 4; ++nf) {
      const __bf16* vp = vptr + (size_t)(nf * 16 + r) * T + k0 + half * 8;
      const v16bf vf = make_frag(vp, vp + 16);
#pragma unroll
      for (int i = 0; i < 8; ++i) o[nf][i] *= alpha[i];
      o[nf] = WMMA_BF16(pa, vf, o[nf]);
    }
  }

  // ---- normalize, write y[b, t, h*HD + d] as bf16 for the out-proj ----
#pragma unroll
  for (int i = 0; i < 8; ++i) {
    const int row = q0 + i + half * 8;
    const float inv = 1.0f / lrow[i];
#pragma unroll
    for (int nf = 0; nf < 4; ++nf)
      yb[((size_t)b * T + row) * C + h * HD + nf * 16 + r] =
          (__bf16)(o[nf][i] * inv);
  }
}

// ---------------------------------------------------------------------------
extern "C" void kernel_launch(void* const* d_in, const int* in_sizes, int n_in,
                              void* d_out, int out_size, void* d_ws, size_t ws_size,
                              hipStream_t stream) {
  constexpr int B = 4, T = 2048, C = 1024, H = 16;
  constexpr int M = B * T;  // 8192

  const float* x   = (const float*)d_in[0];
  const float* Wa  = (const float*)d_in[1];
  const float* Wp  = (const float*)d_in[2];
  const int*   idx = (const int*)d_in[3];
  float* out = (float*)d_out;

  char* ws = (char*)d_ws;
  constexpr size_t SZ_QKV  = (size_t)M * 3 * C * 2;  // 48 MB bf16
  constexpr size_t SZ_HEAD = (size_t)M * C * 2;      // 16 MB bf16
  __bf16* qkvb = (__bf16*)(ws);
  __bf16* qb   = (__bf16*)(ws + SZ_QKV);
  __bf16* kb   = (__bf16*)(ws + SZ_QKV + SZ_HEAD);
  __bf16* vt   = (__bf16*)(ws + SZ_QKV + 2 * SZ_HEAD);
  __bf16* yb   = (__bf16*)(ws + SZ_QKV + 3 * SZ_HEAD);

  // 1) qkv = x @ W_attn   (8192 x 3072 x 1024), bf16 output
  gemm_wmma_kernel<float, __bf16>
      <<<dim3(3 * C / 128, M / 128), 256, 0, stream>>>(x, Wa, qkvb, M, 3 * C, C);

  // 2) RoPE + head-major repack (+ q pre-scale, v transpose)
  rope_pack_kernel<<<(B * T * H * 32) / 256, 256, 0, stream>>>(qkvb, idx, qb, kb, vt);

  // 3) flash attention, one 16-row q tile per wave, 4 waves/block
  attn_kernel<<<dim3(T / 64, B * H), 128, 0, stream>>>(qb, kb, vt, yb);

  // 4) out = y @ W_proj   (8192 x 1024 x 1024), fp32 output, TDM-staged A
  gemm_wmma_kernel<__bf16, float>
      <<<dim3(C / 128, M / 128), 256, 0, stream>>>(yb, Wp, out, M, C, C);
}
